// LLaVA_Alignment_44341242364213
// MI455X (gfx1250) — matrix operand order repaired and checked by reading, compiled, not run on hardware
//
#include <hip/hip_runtime.h>
#include <hip/hip_bf16.h>

// ---------------------------------------------------------------------------
// LLaVA alignment forward on MI455X (gfx1250, wave32).
// Head GEMM (4096x32000x4096) dominates and is HBM-bound: min traffic is
// head_w (262 MB bf16) + logits (524 MB f32) ~= 0.8 GB -> ~35 us @ 23.3 TB/s.
// Strategy: bf16 operands + f32-accumulate WMMA (v_wmma_f32_16x16x32_bf16),
// 128x128 block tiles, K-step 32, DOUBLE-BUFFERED async global->LDS copies
// (ASYNCcnt) so the next tile streams from HBM while the current one feeds
// the WMMA pipe. Grid is M-fastest so each head_w N-strip is L2-resident
// across all 32 M-tiles (head_w read from HBM exactly once).
// ---------------------------------------------------------------------------

typedef __attribute__((ext_vector_type(16))) __bf16          v16bf;
typedef __attribute__((ext_vector_type(8)))  float           v8f;
typedef __attribute__((ext_vector_type(8)))  unsigned short  v8u16;

__device__ __forceinline__ unsigned short f32_to_bf16(float f) {
    unsigned int u = __float_as_uint(f);
    unsigned int r = (u + 0x7FFFu + ((u >> 16) & 1u)) >> 16;   // RNE
    return (unsigned short)r;
}

struct Pair16 { v8u16 lo, hi; };

__device__ __forceinline__ v16bf make_frag(const unsigned short* lds, int off0, int off1) {
    Pair16 p;
    p.lo = *(const v8u16*)(lds + off0);
    p.hi = *(const v8u16*)(lds + off1);
    return __builtin_bit_cast(v16bf, p);
}

// Async 16B global -> LDS copy (CDNA5, tracked with ASYNCcnt).
// Generic LDS pointer low 32 bits == wave-relative LDS address (ISA 10.2).
__device__ __forceinline__ void async_copy_b128(void* lds_ptr, const void* gptr) {
    unsigned lds_off = (unsigned)(unsigned long long)lds_ptr;
    unsigned long long ga = (unsigned long long)gptr;
    asm volatile("global_load_async_to_lds_b128 %0, %1, off"
                 :: "v"(lds_off), "v"(ga) : "memory");
}
__device__ __forceinline__ void async_wait0() {
    asm volatile("s_wait_asynccnt 0x0" ::: "memory");
}

// ---------------------------------------------------------------------------
// fp32 -> bf16 bulk convert, 8 elements per thread iteration (b128 load/store)
// ---------------------------------------------------------------------------
__global__ __launch_bounds__(256)
void cvt_bf16_kernel(const float* __restrict__ in, unsigned short* __restrict__ out, long n8) {
    long i = (long)blockIdx.x * 256 + threadIdx.x;
    long stride = (long)gridDim.x * 256;
    for (; i < n8; i += stride) {
        const float4* p = (const float4*)(in + i * 8);
        float4 x0 = p[0], x1 = p[1];
        v8u16 o;
        o[0] = f32_to_bf16(x0.x); o[1] = f32_to_bf16(x0.y);
        o[2] = f32_to_bf16(x0.z); o[3] = f32_to_bf16(x0.w);
        o[4] = f32_to_bf16(x1.x); o[5] = f32_to_bf16(x1.y);
        o[6] = f32_to_bf16(x1.z); o[7] = f32_to_bf16(x1.w);
        *(v8u16*)(out + i * 8) = o;
    }
}

// ---------------------------------------------------------------------------
// Embedding gather: hidden[b, 576+j, :] = bf16(embed[id, :])
// ---------------------------------------------------------------------------
__global__ __launch_bounds__(256)
void gather_embed_kernel(const int* __restrict__ prompt_ids, const int* __restrict__ answer_ids,
                         const float* __restrict__ embed, unsigned short* __restrict__ hidden) {
    int t = blockIdx.x;            // 0 .. B*448-1
    int b = t / 448, j = t % 448;
    int id = (j < 128) ? prompt_ids[b * 128 + j] : answer_ids[b * 320 + (j - 128)];
    const float* src = embed + (size_t)id * 4096;
    unsigned short* dst = hidden + ((size_t)(b * 1024 + 576 + j)) * 4096;
    for (int i = threadIdx.x; i < 512; i += 256) {
        const float4* p = (const float4*)(src + i * 8);
        float4 x0 = p[0], x1 = p[1];
        v8u16 o;
        o[0] = f32_to_bf16(x0.x); o[1] = f32_to_bf16(x0.y);
        o[2] = f32_to_bf16(x0.z); o[3] = f32_to_bf16(x0.w);
        o[4] = f32_to_bf16(x1.x); o[5] = f32_to_bf16(x1.y);
        o[6] = f32_to_bf16(x1.z); o[7] = f32_to_bf16(x1.w);
        *(v8u16*)(dst + i * 8) = o;
    }
}

// ---------------------------------------------------------------------------
// WMMA GEMM:  C[M,N] = A[M,K] * B[N,K]^T + bias[N]
// A, B row-major bf16; C row-major (bf16 or f32). 256 threads = 8 waves.
// Block tile 128x128; wave tile 64x32 (4x2 fragments of 16x16); K-step 32.
// LDS row stride 40 ushorts (80 B) -> 16B-aligned chunks, spread banks.
// Double-buffered LDS (2 x 20 KB): one s_wait_asynccnt + one barrier per
// K-step; next tile's async copies overlap current tile's WMMA work.
// ---------------------------------------------------------------------------
#define TILE_USH (128 * 40)          // one matrix tile in ushorts (10240 B)
#define BUF_USH  (2 * TILE_USH)      // A tile + B tile per buffer

template <bool OUT_BF16>
__global__ __launch_bounds__(256)
void wmma_gemm_kernel(const unsigned short* __restrict__ A,
                      const unsigned short* __restrict__ Bw,
                      const float* __restrict__ bias,
                      void* __restrict__ Cout,
                      int M, int N, int K, int ldc,
                      size_t aBatchStride, size_t cBatchStride) {
    __shared__ unsigned short smem[2 * BUF_USH];   // [buf][A tile | B tile]

    A = A + (size_t)blockIdx.z * aBatchStride;

    const int tid   = threadIdx.x;
    const int wave  = tid >> 5;
    const int lane  = tid & 31;
    const int rsel  = lane >> 4;     // 0: lanes 0-15, 1: lanes 16-31
    const int mr    = lane & 15;
    const int waveM = wave >> 2;     // 0..1  -> 64-row strip
    const int waveN = wave & 3;      // 0..3  -> 32-col strip
    const int blockM0 = blockIdx.x * 128;
    const int blockN0 = blockIdx.y * 128;

    v8f acc[4][2];
    for (int i = 0; i < 4; ++i)
        for (int j = 0; j < 2; ++j)
            for (int e = 0; e < 8; ++e) acc[i][j][e] = 0.0f;

    // Per-thread tile-fill coordinates: 512 16B chunks per tile, 2 per thread.
    const int r0  = tid >> 1;              // chunk row handled by this thread
    const int cc0 = (tid & 1) * 2;         // chunk cols cc0, cc0+1
    int gra = blockM0 + r0; gra = (gra < M) ? gra : (M - 1);
    int grb = blockN0 + r0; grb = (grb < N) ? grb : (N - 1);
    const unsigned short* gA = A  + (size_t)gra * K + cc0 * 8;
    const unsigned short* gB = Bw + (size_t)grb * K + cc0 * 8;

    // Issue async fill of buffer 0 for k0 = 0.
    #pragma unroll
    for (int c = 0; c < 2; ++c) {
        async_copy_b128(smem + r0 * 40 + (cc0 + c) * 8,            gA + c * 8);
        async_copy_b128(smem + TILE_USH + r0 * 40 + (cc0 + c) * 8, gB + c * 8);
    }

    for (int k0 = 0; k0 < K; k0 += 32) {
        const int cur = (k0 >> 5) & 1;
        unsigned short* lA = smem + cur * BUF_USH;
        unsigned short* lB = lA + TILE_USH;

        async_wait0();       // this wave's outstanding fills (incl. cur) done
        __syncthreads();     // => every wave's fills done; prev buffer free

        // Kick off the next tile into the other buffer; overlaps compute.
        if (k0 + 32 < K) {
            unsigned short* nA = smem + (cur ^ 1) * BUF_USH;
            unsigned short* nB = nA + TILE_USH;
            #pragma unroll
            for (int c = 0; c < 2; ++c) {
                async_copy_b128(nA + r0 * 40 + (cc0 + c) * 8, gA + k0 + 32 + c * 8);
                async_copy_b128(nB + r0 * 40 + (cc0 + c) * 8, gB + k0 + 32 + c * 8);
            }
        }

        v16bf afr[4], bfr[2];
        #pragma unroll
        for (int i = 0; i < 4; ++i) {
            int row = waveM * 64 + i * 16 + mr;            // A: M = lane%16
            int o0 = row * 40 + rsel * 8;                  // K 0-7 / 8-15
            afr[i] = make_frag(lA, o0, o0 + 16);           // + K 16-23 / 24-31
        }
        #pragma unroll
        for (int j = 0; j < 2; ++j) {
            int row = waveN * 32 + j * 16 + mr;            // B: N = lane%16
            int o0 = row * 40 + rsel * 8;                  // K 0-7 / 8-15
            bfr[j] = make_frag(lB, o0, o0 + 16);           // + K 16-23 / 24-31
        }
        #pragma unroll
        for (int i = 0; i < 4; ++i)
            #pragma unroll
            for (int j = 0; j < 2; ++j)
                acc[i][j] = __builtin_amdgcn_wmma_f32_16x16x32_bf16(
                    false, afr[i], false, bfr[j], (short)0, acc[i][j], false, false);
    }

    // Epilogue. C frag layout: VGPR r, lane -> n = lane%16, m = r + 8*(lane/16).
    // N is always a multiple of 128 here; only M can have a ragged last tile.
    if (blockM0 + 128 <= M) {
        #pragma unroll
        for (int i = 0; i < 4; ++i) {
            #pragma unroll
            for (int j = 0; j < 2; ++j) {
                int col = blockN0 + waveN * 32 + j * 16 + mr;
                float bv = bias[col];
                #pragma unroll
                for (int r = 0; r < 8; ++r) {
                    int row = blockM0 + waveM * 64 + i * 16 + rsel * 8 + r;
                    float v = acc[i][j][r] + bv;
                    if (OUT_BF16) {
                        ((unsigned short*)Cout)[(size_t)blockIdx.z * cBatchStride +
                                                (size_t)row * ldc + col] = f32_to_bf16(v);
                    } else {
                        ((float*)Cout)[(size_t)blockIdx.z * cBatchStride +
                                       (size_t)row * ldc + col] = v;
                    }
                }
            }
        }
    } else {
        #pragma unroll
        for (int i = 0; i < 4; ++i) {
            #pragma unroll
            for (int j = 0; j < 2; ++j) {
                int col = blockN0 + waveN * 32 + j * 16 + mr;
                float bv = bias[col];
                #pragma unroll
                for (int r = 0; r < 8; ++r) {
                    int row = blockM0 + waveM * 64 + i * 16 + rsel * 8 + r;
                    if (row < M) {
                        float v = acc[i][j][r] + bv;
                        if (OUT_BF16) {
                            ((unsigned short*)Cout)[(size_t)blockIdx.z * cBatchStride +
                                                    (size_t)row * ldc + col] = f32_to_bf16(v);
                        } else {
                            ((float*)Cout)[(size_t)blockIdx.z * cBatchStride +
                                           (size_t)row * ldc + col] = v;
                        }
                    }
                }
            }
        }
    }
}

// ---------------------------------------------------------------------------
// Per-row NLL: nll[b,a] = logsumexp(logits[b, 703+a, :]) - logits[..., label]
// ---------------------------------------------------------------------------
__global__ __launch_bounds__(256)
void row_nll_kernel(const float* __restrict__ logits, const int* __restrict__ answer_ids,
                    float* __restrict__ nll_out) {
    int a = blockIdx.x;            // 0..319
    int b = blockIdx.y;            // 0..3
    int l = 703 + a;               // shifted position; label = answer_ids[b,a]
    const float* row = logits + ((size_t)(b * 1024 + l)) * 32000;
    int label = answer_ids[b * 320 + a];

    __shared__ float red[256];
    int tid = threadIdx.x;

    float mx = -3.4e38f;
    for (int i = tid; i < 32000; i += 256) mx = fmaxf(mx, row[i]);
    red[tid] = mx; __syncthreads();
    for (int s = 128; s > 0; s >>= 1) {
        if (tid < s) red[tid] = fmaxf(red[tid], red[tid + s]);
        __syncthreads();
    }
    mx = red[0]; __syncthreads();

    float sum = 0.0f;
    for (int i = tid; i < 32000; i += 256) sum += __expf(row[i] - mx);
    red[tid] = sum; __syncthreads();
    for (int s = 128; s > 0; s >>= 1) {
        if (tid < s) red[tid] += red[tid + s];
        __syncthreads();
    }
    if (tid == 0) nll_out[b * 320 + a] = (mx + __logf(red[0])) - row[label];
}

// Deterministic final reduction (fixed order, no float atomics).
__global__ __launch_bounds__(256)
void loss_reduce_kernel(const float* __restrict__ nll, float* __restrict__ out) {
    __shared__ float red[256];
    int tid = threadIdx.x;
    float s = 0.0f;
    for (int i = tid; i < 1280; i += 256) s += nll[i];
    red[tid] = s; __syncthreads();
    for (int st = 128; st > 0; st >>= 1) {
        if (tid < st) red[tid] += red[tid + st];
        __syncthreads();
    }
    if (tid == 0) out[0] = red[0] / 1280.0f;
}

// ---------------------------------------------------------------------------
extern "C" void kernel_launch(void* const* d_in, const int* in_sizes, int n_in,
                              void* d_out, int out_size, void* d_ws, size_t ws_size,
                              hipStream_t stream) {
    const float* image_features = (const float*)d_in[0];  // (4,576,1024)
    const int*   prompt_ids     = (const int*)d_in[1];    // (4,128)
    const int*   answer_ids     = (const int*)d_in[2];    // (4,320)
    const float* proj_w         = (const float*)d_in[3];  // (4096,1024)
    const float* proj_b         = (const float*)d_in[4];  // (4096,)
    const float* embed          = (const float*)d_in[5];  // (32000,4096)
    const float* head_w         = (const float*)d_in[6];  // (32000,4096)
    const float* head_b         = (const float*)d_in[7];  // (32000,)

    float* out    = (float*)d_out;
    float* logits = out + 1;                              // (4,1024,32000) f32

    // Workspace layout (bf16 operands + nll scratch), ~309 MB.
    unsigned short* imgbf    = (unsigned short*)d_ws;                 // 2304*1024
    unsigned short* projwbf  = imgbf   + (size_t)2304 * 1024;         // 4096*1024
    unsigned short* hiddenbf = projwbf + (size_t)4096 * 1024;         // 4096*4096
    unsigned short* headwbf  = hiddenbf + (size_t)4096 * 4096;        // 32000*4096
    float*          nll      = (float*)(headwbf + (size_t)32000 * 4096); // 1280

    // 1) fp32 -> bf16 conversions (one-pass streams).
    cvt_bf16_kernel<<<2048, 256, 0, stream>>>(image_features, imgbf, (long)2304 * 1024 / 8);
    cvt_bf16_kernel<<<4096, 256, 0, stream>>>(proj_w, projwbf, (long)4096 * 1024 / 8);
    cvt_bf16_kernel<<<16384, 256, 0, stream>>>(head_w, headwbf, (long)32000 * 4096 / 8);

    // 2) Projection GEMM per batch: image rows of hidden (bf16 out).
    //    M=576, N=4096, K=1024; grid.x = M tiles, grid.y = N tiles, z = batch.
    {
        dim3 grid((576 + 127) / 128, 4096 / 128, 4);
        wmma_gemm_kernel<true><<<grid, 256, 0, stream>>>(
            imgbf, projwbf, proj_b, (void*)hiddenbf,
            576, 4096, 1024, 4096,
            (size_t)576 * 1024, (size_t)1024 * 4096);
    }

    // 3) Prompt/answer embedding gather into bf16 hidden rows 576..1023.
    gather_embed_kernel<<<4 * 448, 256, 0, stream>>>(prompt_ids, answer_ids, embed, hiddenbf);

    // 4) Head GEMM: logits = hidden(4096x4096) * head_w(32000x4096)^T + head_b.
    //    grid.x = 32 M tiles (fast), grid.y = 250 N strips -> head_w streamed
    //    from HBM once, hidden (32 MB bf16) L2-resident.
    {
        dim3 grid(4096 / 128, 32000 / 128, 1);
        wmma_gemm_kernel<false><<<grid, 256, 0, stream>>>(
            hiddenbf, headwbf, head_b, (void*)logits,
            4096, 32000, 4096, 32000, 0, 0);
    }

    // 5) Loss: per-row logsumexp/NLL on the 4*320 valid shifted positions,
    //    then deterministic reduce into d_out[0].
    row_nll_kernel<<<dim3(320, 4), 256, 0, stream>>>(logits, answer_ids, nll);
    loss_reduce_kernel<<<1, 256, 0, stream>>>(nll, out);
}
